// LowRankSoftmaxAttentionBlock_75041668596014
// MI455X (gfx1250) — matrix operand order repaired and checked by reading, not compiled, not run on hardware
//
#include <hip/hip_runtime.h>
#include <hip/hip_bf16.h>
#include <math.h>

typedef __attribute__((ext_vector_type(16))) _Float16 v16h;
typedef __attribute__((ext_vector_type(8)))  float    v8f;

#define D_MODEL 256
#define NTOK    4096
#define BATCH   8
#define NB      32
#define NKEY    2048
#define LN_EPS  1e-5f

__device__ __forceinline__ v8f wmma_f16(v16h a, v16h b, v8f c) {
  return __builtin_amdgcn_wmma_f32_16x16x32_f16(false, a, false, b, (short)0, c, false, false);
}

// ---- WMMA fragment helpers (wave32; A/B k contiguous per lane half, C per ISA 7.12.2) ----

// A fragment: 16x32 f16 tile at (m0,k0) of row-major [.,ld]
// lanes 0-15: row m0+lane, K {0..15}; lanes 16-31: row m0+lane-16, K {16..31}
__device__ __forceinline__ v16h load_a_frag(const _Float16* base, int ld, int m0, int k0) {
  int lane = threadIdx.x & 31;
  int hi   = lane >> 4;
  const _Float16* p = base + (size_t)(m0 + (lane & 15)) * ld + k0 + hi * 16;
  v16h a;
#pragma unroll
  for (int i = 0; i < 8; ++i) { a[i] = p[i]; a[8 + i] = p[8 + i]; }
  return a;
}

// B fragment from row-major [K,ld] (used only for small LDS-side GEMMs)
__device__ __forceinline__ v16h load_b_frag_rm(const _Float16* base, int ld, int k0, int n0) {
  int lane = threadIdx.x & 31;
  int n  = n0 + (lane & 15);
  int kb = k0 + (lane >> 4) * 16;
  v16h b;
#pragma unroll
  for (int r = 0; r < 8; ++r) {
    b[2 * r]     = base[(size_t)(kb + 2 * r)     * ld + n];
    b[2 * r + 1] = base[(size_t)(kb + 2 * r + 1) * ld + n];
  }
  return b;
}

// ---- Swizzled B storage ----
// K×N matrix tiled into 32×16 fragments (512 halves = 1KB each), tile (kt,nt)
// at linear tile index (nt*ktiles + kt). Within a tile lane l owns 16 contiguous
// halves at [l*16], value i = element(k = kt*32 + (l>>4)*16 + i, n = nt*16 + (l&15))
// — the same contiguous k order load_a_frag uses, so a fragment load is one 32-byte vector load.
__device__ __forceinline__ size_t b_swz_idx(int ktiles, int k, int n) {
  return ((size_t)((n >> 4) * ktiles + (k >> 5)) << 9)
       + ((((k >> 4) & 1) * 16 + (n & 15)) << 4) + (k & 15);
}

__device__ __forceinline__ v16h load_b_frag_swz(const _Float16* base, int ktiles, int kt, int nt) {
  int lane = threadIdx.x & 31;
  const v16h* p = (const v16h*)(base + (((size_t)(nt * ktiles + kt)) << 9) + (lane << 4));
  return *p;
}

// C fragment stores: VGPR r -> row m0+r (lanes 0-15) / m0+8+r (lanes 16-31)
__device__ __forceinline__ void store_c_f32(float* dst, int ld, int m0, int n0, v8f c, float mul) {
  int lane = threadIdx.x & 31;
  int n  = n0 + (lane & 15);
  int mb = m0 + (lane >> 4) * 8;
#pragma unroll
  for (int r = 0; r < 8; ++r) dst[(size_t)(mb + r) * ld + n] = c[r] * mul;
}

__device__ __forceinline__ void store_c_f16(_Float16* dst, int ld, int m0, int n0, v8f c) {
  int lane = threadIdx.x & 31;
  int n  = n0 + (lane & 15);
  int mb = m0 + (lane >> 4) * 8;
#pragma unroll
  for (int r = 0; r < 8; ++r) dst[(size_t)(mb + r) * ld + n] = (_Float16)c[r];
}

// ---- Kernel 1: weight precision/layout prep (swizzled B layouts) ----
__global__ void prep_weights(const float* Wq, const float* Wk, const float* Wv, const float* Wo,
                             const float* E, const float* F,
                             _Float16* WqS, _Float16* WkS, _Float16* WvS, _Float16* WoS,
                             _Float16* Ef, _Float16* Ff) {
  int idx = blockIdx.x * blockDim.x + threadIdx.x;
  if (idx < 65536) {
    // dst-linear decompose: 256x256, ktiles = 8
    int tile = idx >> 9, within = idx & 511;
    int lane = within >> 4, i = within & 15;
    int nt = tile >> 3, kt = tile & 7;
    int k = kt * 32 + (lane >> 4) * 16 + i;   // input dim d
    int n = nt * 16 + (lane & 15);            // output dim h
    int src = n * 256 + k;                    // B[k][n] = W[n][k]
    WqS[idx] = (_Float16)Wq[src];
    WkS[idx] = (_Float16)Wk[src];
    WvS[idx] = (_Float16)Wv[src];
    WoS[idx] = (_Float16)Wo[src];
  }
  if (idx < 8192) { Ef[idx] = (_Float16)E[idx]; Ff[idx] = (_Float16)F[idx]; }
}

// ---- Kernel 2: LayerNorm + QKV GEMMs + L2norm + block projections ----
// one workgroup per (batch, 128-token block); 256 threads = 8 waves
__global__ void ln_qkv_proj(const float* __restrict__ tokens,
                            const _Float16* __restrict__ WqS, const _Float16* __restrict__ WkS,
                            const _Float16* __restrict__ WvS,
                            const _Float16* __restrict__ Ef, const _Float16* __restrict__ Ff,
                            const float* __restrict__ g1, const float* __restrict__ b1,
                            _Float16* __restrict__ Qn,    // [B][4096][256] f16 row-major
                            _Float16* __restrict__ KTs,   // [B] swizzled B: K=256(h) x N=2048(key)
                            _Float16* __restrict__ Vps) { // [B] swizzled B: K=2048(key) x N=256(v)
  extern __shared__ char smem[];
  _Float16* tn    = (_Float16*)smem;                 // 128*256 f16 = 64 KB
  _Float16* buf   = (_Float16*)(smem + 65536);       // 128*256 f16 = 64 KB
  float*    rstat = (float*)(smem + 131072);         // 128 f32

  int b   = blockIdx.x >> 5;
  int blk = blockIdx.x & 31;
  int t = threadIdx.x, wave = t >> 5, lane = t & 31;
  const float* tokbase = tokens + ((size_t)(b * NTOK + blk * 128)) * 256;

  // LayerNorm -> tn (f16), 2 threads per row
  {
    int row = t >> 1, half = t & 1;
    const float* tr = tokbase + (size_t)row * 256;
    float s = 0.f, ss = 0.f;
    for (int i = 0; i < 128; ++i) { float v = tr[half * 128 + i]; s += v; ss += v * v; }
    s  += __shfl_xor(s, 1, 32);
    ss += __shfl_xor(ss, 1, 32);
    float mean = s * (1.f / 256.f);
    float var  = ss * (1.f / 256.f) - mean * mean;
    float inv  = rsqrtf(var + LN_EPS);
    for (int i = 0; i < 128; ++i) {
      int col = half * 128 + i;
      tn[row * 256 + col] = (_Float16)((tr[col] - mean) * inv * g1[col] + b1[col]);
    }
  }
  __syncthreads();

  for (int which = 0; which < 3; ++which) {      // 0:Q 1:K 2:V
    const _Float16* Wt = (which == 0) ? WqS : ((which == 1) ? WkS : WvS);
    int m0 = wave * 16;
    float ssq[8];
#pragma unroll
    for (int r = 0; r < 8; ++r) ssq[r] = 0.f;
    // buf[128x256] = tn @ W
    for (int n = 0; n < 16; ++n) {
      v8f c = {};
      for (int k0 = 0; k0 < 8; ++k0) {
        v16h a  = load_a_frag(tn, 256, m0, k0 * 32);
        v16h bf = load_b_frag_swz(Wt, 8, k0, n);
        c = wmma_f16(a, bf, c);
      }
#pragma unroll
      for (int r = 0; r < 8; ++r) ssq[r] += c[r] * c[r];
      store_c_f16(buf, 256, m0, n * 16, c);
    }
    if (which < 2) {                             // per-row L2 norm factors
#pragma unroll
      for (int off = 1; off < 16; off <<= 1)
#pragma unroll
        for (int r = 0; r < 8; ++r) ssq[r] += __shfl_xor(ssq[r], off, 32);
      if (lane == 0 || lane == 16) {
        int rb = m0 + (lane >> 4) * 8;
#pragma unroll
        for (int r = 0; r < 8; ++r)
          rstat[rb + r] = 1.f / fmaxf(sqrtf(ssq[r]), 1e-12f);
      }
    }
    __syncthreads();

    if (which == 0) {
      _Float16* Qb = Qn + ((size_t)(b * NTOK + blk * 128)) * 256;
      for (int e = t; e < 128 * 256; e += 256) {
        int row = e >> 8;
        Qb[e] = (_Float16)((float)buf[e] * rstat[row]);
      }
      __syncthreads();
      continue;
    }
    if (which == 1) {                            // scale K rows in place
      for (int e = t; e < 128 * 256; e += 256) {
        int row = e >> 8;
        buf[e] = (_Float16)((float)buf[e] * rstat[row]);
      }
      __syncthreads();
    }
    // projection: [64x256] = E/F (64x128) @ buf (128x256)
    const _Float16* Af = (which == 1) ? Ef : Ff;
    int pm  = (wave & 3) * 16;
    int nb0 = (wave >> 2) * 8;
    int hb  = (lane >> 4) * 8;
    v8f accp[8];
    v8f zf = {};
#pragma unroll
    for (int j = 0; j < 8; ++j) accp[j] = zf;
    for (int k0 = 0; k0 < 4; ++k0) {
      v16h a = load_a_frag(Af, 128, pm, k0 * 32);
      for (int j = 0; j < 8; ++j) {
        v16h bf = load_b_frag_rm(buf, 256, k0 * 32, (nb0 + j) * 16);
        accp[j] = wmma_f16(a, bf, accp[j]);
      }
    }
    if (which == 1) {
      // K_proj -> swizzled B matrix [K=h=256][N=key=2048], ktiles=8
      _Float16* KTb = KTs + (size_t)b * 256 * NKEY;
      for (int j = 0; j < 8; ++j) {
        int h   = (nb0 + j) * 16 + (lane & 15);  // k-dim of score-B
        int kb  = blk * 64 + pm + hb;            // n-dim (key) base
#pragma unroll
        for (int r = 0; r < 8; ++r)
          KTb[b_swz_idx(8, h, kb + r)] = (_Float16)accp[j][r];
      }
    } else {
      // V_proj -> swizzled B matrix [K=key=2048][N=v=256], ktiles=64
      _Float16* Vb = Vps + (size_t)b * NKEY * 256;
      for (int j = 0; j < 8; ++j) {
        int v   = (nb0 + j) * 16 + (lane & 15);  // n-dim
        int key = blk * 64 + pm + hb;            // k-dim; r gives consecutive halves
        union { _Float16 h8[8]; uint4 u; } pk;
#pragma unroll
        for (int r = 0; r < 8; ++r) pk.h8[r] = (_Float16)accp[j][r];
        *(uint4*)(Vb + b_swz_idx(64, key, v)) = pk.u;
      }
    }
    __syncthreads();
  }
}

// ---- Kernel 3: flash-style attention + output projection + residual LN ----
// one workgroup per (batch, 64-query tile); 256 threads = 8 waves
__global__ void attention(const float* __restrict__ tokens,
                          const _Float16* __restrict__ Qn, const _Float16* __restrict__ KTs,
                          const _Float16* __restrict__ Vps, const _Float16* __restrict__ WoS,
                          const float* __restrict__ g2, const float* __restrict__ b2,
                          const float* __restrict__ scale_p, float* __restrict__ out) {
  extern __shared__ char smem[];
  _Float16* Qt      = (_Float16*)smem;               // 64*256 f16 = 32768
  float*    scoresF = (float*)(smem + 32768);        // 64*128 f32 = 32768
  _Float16* attnh   = (_Float16*)(smem + 32768);     // alias (used after chunk loop)
  _Float16* ph      = (_Float16*)(smem + 65536);     // 64*128 f16 = 16384
  float*    outF    = (float*)(smem + 81920);        // 64*256 f32 = 65536
  float*    mrow    = (float*)(smem + 147456);       // 64
  float*    lrow    = (float*)(smem + 147712);       // 64
  float*    corr    = (float*)(smem + 147968);       // 64  (total 148224 B)

  int b  = blockIdx.x >> 6;
  int rt = blockIdx.x & 63;
  int t = threadIdx.x, wave = t >> 5, lane = t & 31;
  const _Float16* KTb = KTs + (size_t)b * 256 * NKEY;
  const _Float16* Vpb = Vps + (size_t)b * NKEY * 256;

  float s = log1pf(expf(scale_p[0])) * 0.0625f;      // softplus(scale)/sqrt(256)

  { // stage Q tile into LDS (b128 copies)
    const uint4* src = (const uint4*)(Qn + ((size_t)(b * NTOK + rt * 64)) * 256);
    uint4* dst = (uint4*)Qt;
    for (int i = t; i < 64 * 256 / 8; i += 256) dst[i] = src[i];
  }
  if (t < 64) { mrow[t] = -1e30f; lrow[t] = 0.f; }
  __syncthreads();

  int am  = (wave & 3) * 16;                         // accumulator row tile
  int an0 = (wave >> 2) * 128;                       // accumulator col base
  int hb  = (lane >> 4) * 8;
  v8f acc[8];
  v8f zf = {};
#pragma unroll
  for (int j = 0; j < 8; ++j) acc[j] = zf;

  for (int ch = 0; ch < 16; ++ch) {
    // prefetch next chunk's KT/Vp slabs (64 KB each) into cache
    if (ch < 15) {
      const char* kpre = (const char*)(KTb + (((size_t)((ch + 1) * 8) * 8) << 9));
      const char* vpre = (const char*)(Vpb + (((size_t)((ch + 1) * 4)) << 9));
      __builtin_prefetch(kpre + t * 256, 0, 3);
      __builtin_prefetch(kpre + t * 256 + 128, 0, 3);
      __builtin_prefetch(vpre + t * 256, 0, 3);
      __builtin_prefetch(vpre + t * 256 + 128, 0, 3);
    }

    // scores chunk: [64x128] = Q(64x256) @ K_projT(256 x 128 keys)
    v8f cs[4];
#pragma unroll
    for (int mi = 0; mi < 4; ++mi) cs[mi] = zf;
    for (int k0 = 0; k0 < 8; ++k0) {
      v16h bf = load_b_frag_swz(KTb, 8, k0, ch * 8 + wave);
      for (int mi = 0; mi < 4; ++mi) {
        v16h a = load_a_frag(Qt, 256, mi * 16, k0 * 32);
        cs[mi] = wmma_f16(a, bf, cs[mi]);
      }
    }
    for (int mi = 0; mi < 4; ++mi)
      store_c_f32(scoresF, 128, mi * 16, wave * 16, cs[mi], s);
    __syncthreads();

    // online softmax state update: 4 threads per query row
    {
      int row = t >> 2, seg = t & 3;
      const float* srow = scoresF + row * 128 + seg * 32;
      float mx = -1e30f;
      for (int i = 0; i < 32; ++i) mx = fmaxf(mx, srow[i]);
      mx = fmaxf(mx, __shfl_xor(mx, 1, 32));
      mx = fmaxf(mx, __shfl_xor(mx, 2, 32));
      float mo = mrow[row];
      mx = fmaxf(mx, mo);
      float sum = 0.f;
      _Float16* prow = ph + row * 128 + seg * 32;
      for (int i = 0; i < 32; ++i) {
        float e = expf(srow[i] - mx);
        sum += e;
        prow[i] = (_Float16)e;
      }
      sum += __shfl_xor(sum, 1, 32);
      sum += __shfl_xor(sum, 2, 32);
      if (seg == 0) {
        float cr = expf(mo - mx);
        lrow[row] = lrow[row] * cr + sum;
        mrow[row] = mx;
        corr[row] = cr;
      }
    }
    __syncthreads();

    // rescale running accumulator, then acc += P @ V_proj chunk
    {
      float f[8];
#pragma unroll
      for (int r = 0; r < 8; ++r) f[r] = corr[am + hb + r];
#pragma unroll
      for (int j = 0; j < 8; ++j)
#pragma unroll
        for (int r = 0; r < 8; ++r) acc[j][r] *= f[r];
    }
    for (int k0 = 0; k0 < 4; ++k0) {
      v16h a = load_a_frag(ph, 128, am, k0 * 32);
      for (int j = 0; j < 8; ++j) {
        v16h bf = load_b_frag_swz(Vpb, 64, ch * 4 + k0, (an0 >> 4) + j);
        acc[j] = wmma_f16(a, bf, acc[j]);
      }
    }
    __syncthreads();
  }

  // finalize: attn = acc / l  -> f16 LDS (aliases scoresF region)
  {
    float f[8];
#pragma unroll
    for (int r = 0; r < 8; ++r) f[r] = 1.f / lrow[am + hb + r];
    for (int j = 0; j < 8; ++j) {
      int n  = an0 + j * 16 + (lane & 15);
      int mb = am + hb;
#pragma unroll
      for (int r = 0; r < 8; ++r)
        attnh[(mb + r) * 256 + n] = (_Float16)(acc[j][r] * f[r]);
    }
  }
  __syncthreads();

  // output projection: outF = attn @ Wo^T
  v8f co[8];
#pragma unroll
  for (int j = 0; j < 8; ++j) co[j] = zf;
  for (int k0 = 0; k0 < 8; ++k0) {
    v16h a = load_a_frag(attnh, 256, am, k0 * 32);
    for (int j = 0; j < 8; ++j) {
      v16h bf = load_b_frag_swz(WoS, 8, k0, (an0 >> 4) + j);
      co[j] = wmma_f16(a, bf, co[j]);
    }
  }
  for (int j = 0; j < 8; ++j)
    store_c_f32(outF, 256, am, an0 + j * 16, co[j], 1.0f);
  __syncthreads();

  // residual + final LayerNorm: 4 threads per row
  {
    int row = t >> 2, seg = t & 3;
    size_t rowg = (size_t)(b * NTOK + rt * 64 + row) * 256;
    const float* tr = tokens + rowg + seg * 64;
    float* orow = outF + row * 256 + seg * 64;
    float sum = 0.f, ssum = 0.f;
    for (int i = 0; i < 64; ++i) {
      float v = tr[i] + 0.1f * orow[i];
      orow[i] = v; sum += v; ssum += v * v;
    }
    sum  += __shfl_xor(sum, 1, 32);
    sum  += __shfl_xor(sum, 2, 32);
    ssum += __shfl_xor(ssum, 1, 32);
    ssum += __shfl_xor(ssum, 2, 32);
    float mean = sum * (1.f / 256.f);
    float var  = ssum * (1.f / 256.f) - mean * mean;
    float inv  = rsqrtf(var + LN_EPS);
    float* dst = out + rowg + seg * 64;
    const float* g2s = g2 + seg * 64;
    const float* b2s = b2 + seg * 64;
    for (int i = 0; i < 64; ++i)
      dst[i] = (orow[i] - mean) * inv * g2s[i] + b2s[i];
  }
}

extern "C" void kernel_launch(void* const* d_in, const int* in_sizes, int n_in,
                              void* d_out, int out_size, void* d_ws, size_t ws_size,
                              hipStream_t stream) {
  (void)in_sizes; (void)n_in; (void)out_size; (void)ws_size;
  const float* tokens = (const float*)d_in[0];
  const float* Wq  = (const float*)d_in[1];
  const float* Wk  = (const float*)d_in[2];
  const float* Wv  = (const float*)d_in[3];
  const float* Wo  = (const float*)d_in[4];
  const float* E   = (const float*)d_in[5];
  const float* F   = (const float*)d_in[6];
  const float* g1  = (const float*)d_in[7];
  const float* b1  = (const float*)d_in[8];
  const float* g2  = (const float*)d_in[9];
  const float* b2  = (const float*)d_in[10];
  const float* sc  = (const float*)d_in[11];
  float* out = (float*)d_out;

  _Float16* wsh = (_Float16*)d_ws;
  _Float16* Qn  = wsh;                        // 8*4096*256     (row-major f16)
  _Float16* KTs = wsh + (size_t)8388608;      // 8*256*2048     (swizzled B)
  _Float16* Vps = wsh + (size_t)12582912;     // 8*2048*256     (swizzled B)
  _Float16* WqS = wsh + (size_t)16777216;     // 256*256 each   (swizzled B)
  _Float16* WkS = WqS + 65536;
  _Float16* WvS = WkS + 65536;
  _Float16* WoS = WvS + 65536;
  _Float16* Ef  = WoS + 65536;                // 64*128 row-major (A operand)
  _Float16* Ff  = Ef + 8192;

  prep_weights<<<256, 256, 0, stream>>>(Wq, Wk, Wv, Wo, E, F, WqS, WkS, WvS, WoS, Ef, Ff);
  ln_qkv_proj<<<BATCH * NB, 256, 131584, stream>>>(tokens, WqS, WkS, WvS, Ef, Ff, g1, b1,
                                                   Qn, KTs, Vps);
  attention<<<BATCH * (NTOK / 64), 256, 148224, stream>>>(tokens, Qn, KTs, Vps, WoS,
                                                          g2, b2, sc, out);
}
